// EncoderAttention_31275951849956
// MI455X (gfx1250) — matrix operand order repaired and checked by reading, not compiled, not run on hardware
//
#include <hip/hip_runtime.h>
#include <hip/hip_bf16.h>
#include <stdint.h>

// Problem constants (reference: B=2, S=2048, H=1024, NH=16, HD=64, block_size=4)
#define BB   2
#define SS   2048
#define HH   1024
#define NHH  16
#define HDD  64
#define MM   (BB * SS)          // 4096 rows

typedef __attribute__((ext_vector_type(16))) __bf16 v16bf;
typedef __attribute__((ext_vector_type(8)))  float  v8f;

struct __align__(16) U4 { unsigned int x, y, z, w; };
union ABPack { v16bf v; U4 u[2]; };

__device__ __forceinline__ unsigned short f2bf(float f) {
    unsigned int u = __float_as_uint(f);
    unsigned int r = u + 0x7fffu + ((u >> 16) & 1u);   // round-to-nearest-even
    return (unsigned short)(r >> 16);
}

__device__ __forceinline__ v8f v8f_zero() {
    v8f z = {0.f, 0.f, 0.f, 0.f, 0.f, 0.f, 0.f, 0.f};
    return z;
}

// ---------------------------------------------------------------------------
// f32 -> bf16 elementwise convert
// ---------------------------------------------------------------------------
__global__ __launch_bounds__(256)
void cvt_bf16_kernel(const float* __restrict__ in, unsigned short* __restrict__ out, int n) {
    int i = blockIdx.x * 256 + threadIdx.x;
    if (i < n) out[i] = f2bf(in[i]);
}

// ---------------------------------------------------------------------------
// C(MxN,f32) = A(MxK,bf16 row-major) * B(NxK,bf16 row-major)^T
// Block = 4 waves sharing one 64-wide N panel; the shared B tile (64x32 bf16,
// 4 KB) is staged into LDS with async-to-LDS copies (ASYNCcnt), double
// buffered, so each block reads B once instead of 4x.  One wave computes a
// 16x64 tile: 4 f32 accumulators, k-steps of 32, 4 WMMAs per step.  All four
// B sub-tiles are pulled from LDS before the WMMA burst so the ds_loads can
// be clause-grouped and the WMMAs issue back-to-back.
// ---------------------------------------------------------------------------
__global__ __launch_bounds__(128)
void gemm_bf16_kernel(const unsigned short* __restrict__ A,
                      const unsigned short* __restrict__ Bw,
                      float* __restrict__ C, int M, int N, int K) {
    __shared__ __align__(16) unsigned short Bsh[2][64 * 32];   // 8 KB double buffer

    int lane = threadIdx.x & 31;
    int wave = threadIdx.x >> 5;
    int hf = lane >> 4;        // half-wave (0/1)
    int lm = lane & 15;
    int m0 = (blockIdx.y * 4 + wave) * 16;
    int n0 = blockIdx.x * 64;

    // async fill of one 64x32 B panel into LDS buffer `buf` (2 instr / wave)
    auto fill = [&](int buf, int k0) {
        int t = threadIdx.x;
        int nrow = t >> 1;                    // 0..63
        int hb = t & 1;                       // which 16-half segment
        unsigned lds_addr = (unsigned)(uintptr_t)&Bsh[buf][nrow * 32 + 16 * hb];
        const unsigned short* g = Bw + (size_t)(n0 + nrow) * K + k0 + 16 * hb;
        asm volatile("global_load_async_to_lds_b128 %0, %1, off"
                     :: "v"(lds_addr), "v"(g) : "memory");
        asm volatile("global_load_async_to_lds_b128 %0, %1, off offset:16"
                     :: "v"(lds_addr), "v"(g) : "memory");
    };

    v8f acc[4] = {v8f_zero(), v8f_zero(), v8f_zero(), v8f_zero()};
    const unsigned short* arow = A + (size_t)(m0 + lm) * K;

    int nk = K >> 5;                          // k-steps of 32
    fill(0, 0);
    for (int i = 0; i < nk; ++i) {
        int k0 = i << 5;
        int buf = i & 1;
        if (i + 1 < nk) {
            fill(buf ^ 1, k0 + 32);
            asm volatile("s_wait_asynccnt 0x2" ::: "memory");  // current buf landed
        } else {
            asm volatile("s_wait_asynccnt 0x0" ::: "memory");
        }
        __syncthreads();                      // current buf visible to all waves

        // A 16x32 bf16 layout: k = k0+16*hf+e
        ABPack ap;
        ap.u[0] = *(const U4*)(arow + k0 + 16 * hf);
        ap.u[1] = *(const U4*)(arow + k0 + 16 * hf + 8);

        // pull all four B sub-tiles first (8 ds_load_b128, clause-groupable)
        ABPack bp[4];
#pragma unroll
        for (int t = 0; t < 4; ++t) {
            // B 32x16 bf16 layout: col n = lm, k = 16*hf + e (16 contiguous in LDS)
            const unsigned short* brow = &Bsh[buf][(t * 16 + lm) * 32 + 16 * hf];
            bp[t].u[0] = *(const U4*)(brow);
            bp[t].u[1] = *(const U4*)(brow + 8);
        }
        // then the WMMA burst, back-to-back
#pragma unroll
        for (int t = 0; t < 4; ++t) {
            acc[t] = __builtin_amdgcn_wmma_f32_16x16x32_bf16(
                false, ap.v, false, bp[t].v, (short)0, acc[t], false, false);
        }
        __syncthreads();                      // all waves done with buf before refill
    }
#pragma unroll
    for (int t = 0; t < 4; ++t)
#pragma unroll
        for (int r = 0; r < 8; ++r) {
            int row = m0 + 8 * hf + r;                 // C layout: M = vgpr + 8*half
            C[(size_t)row * N + n0 + t * 16 + lm] = acc[t][r];
        }
}

// ---------------------------------------------------------------------------
// gate_sig[row, n] = sigmoid( dot(x[row,:], Wgate[n,:]) ), one wave per (row,n)
// ---------------------------------------------------------------------------
__global__ __launch_bounds__(128)
void gate_kernel(const float* __restrict__ x, const float* __restrict__ Wgate,
                 float* __restrict__ gate_sig) {
    int gid = blockIdx.x * 4 + (threadIdx.x >> 5);    // 0 .. MM*NHH-1
    int lane = threadIdx.x & 31;
    int row = gid >> 4;
    int n = gid & 15;
    const float* xr = x + (size_t)row * HH;
    const float* wr = Wgate + (size_t)n * HH;
    float s = 0.f;
    for (int i = lane; i < HH; i += 32) s += xr[i] * wr[i];
#pragma unroll
    for (int m = 16; m >= 1; m >>= 1) s += __shfl_xor(s, m, 32);
    if (lane == 0) gate_sig[(size_t)row * NHH + n] = 1.f / (1.f + __expf(-s));
}

// ---------------------------------------------------------------------------
// RoPE for q,k; sigmoid key-gating blend; layout change to (b,h,s,d) bf16 and
// V transposed to (b,h,d,s) bf16.  One wave per (b,h,s); lane owns d and d+32.
// ---------------------------------------------------------------------------
__global__ __launch_bounds__(128)
void rope_gate_kernel(const float* __restrict__ qf, const float* __restrict__ kf,
                      const float* __restrict__ vf, const float* __restrict__ cosT,
                      const float* __restrict__ sinT, const float* __restrict__ Wg,
                      unsigned short* __restrict__ qb, unsigned short* __restrict__ kb,
                      unsigned short* __restrict__ vtb) {
    int gid = blockIdx.x * 4 + (threadIdx.x >> 5);    // (b*NH+h)*S + s
    int lane = threadIdx.x & 31;
    int s = gid & (SS - 1);
    int bh = gid >> 11;
    int b = bh >> 4, h = bh & 15;

    size_t row = ((size_t)(b * SS + s)) * HH + h * HDD;   // (b,s,h,d) f32
    float cl = cosT[s * HDD + lane],      ch = cosT[s * HDD + 32 + lane];
    float sl = sinT[s * HDD + lane],      sh = sinT[s * HDD + 32 + lane];

    // ---- Q RoPE ----
    float qlo = qf[row + lane], qhi = qf[row + 32 + lane];
    float qr_lo = qlo * cl - qhi * sl;
    float qr_hi = qhi * ch + qlo * sh;
    size_t orow = ((size_t)bh * SS + s) * HDD;            // (b,h,s,d) bf16
    qb[orow + lane]      = f2bf(qr_lo);
    qb[orow + 32 + lane] = f2bf(qr_hi);

    // ---- K RoPE + one-step sigmoid gate blend with predecessor ----
    float klo = kf[row + lane], khi = kf[row + 32 + lane];
    float kr_lo = klo * cl - khi * sl;
    float kr_hi = khi * ch + klo * sh;
    float out_lo = kr_lo, out_hi = kr_hi;
    if (s > 0) {
        size_t prow = row - HH;                            // (b, s-1, h, d)
        float c2l = cosT[(s - 1) * HDD + lane], c2h = cosT[(s - 1) * HDD + 32 + lane];
        float s2l = sinT[(s - 1) * HDD + lane], s2h = sinT[(s - 1) * HDD + 32 + lane];
        float plo = kf[prow + lane], phi = kf[prow + 32 + lane];
        float pr_lo = plo * c2l - phi * s2l;
        float pr_hi = phi * c2h + plo * s2h;
        float part = pr_lo * Wg[lane] + pr_hi * Wg[32 + lane];
#pragma unroll
        for (int m = 16; m >= 1; m >>= 1) part += __shfl_xor(part, m, 32);
        float g = 1.f / (1.f + __expf(-part));
        float w = 2.f * g - 1.f;
        out_lo += w * pr_lo;
        out_hi += w * pr_hi;
    }
    kb[orow + lane]      = f2bf(out_lo);
    kb[orow + 32 + lane] = f2bf(out_hi);

    // ---- V transpose to (b,h,d,s) ----
    float vlo = vf[row + lane], vhi = vf[row + 32 + lane];
    vtb[((size_t)bh * HDD + lane) * SS + s]      = f2bf(vlo);
    vtb[((size_t)bh * HDD + 32 + lane) * SS + s] = f2bf(vhi);
}

// ---------------------------------------------------------------------------
// Flash attention, block-causal mask (block=4).  One wave per (b,h, 16-query
// tile).  Keys processed 32 at a time: 4 QK^T wmmas + 4 PV wmmas per chunk.
// Epilogue applies per-head sigmoid gate and writes bf16 (b,s,h*d).
// ---------------------------------------------------------------------------
__global__ __launch_bounds__(128)
void attn_kernel(const unsigned short* __restrict__ qb,
                 const unsigned short* __restrict__ kb,
                 const unsigned short* __restrict__ vtb,
                 const float* __restrict__ gate_sig,
                 unsigned short* __restrict__ ob) {
    __shared__ __align__(16) unsigned short pshare[4 * 16 * 32];   // P staging per wave
    int lane = threadIdx.x & 31;
    int wave = threadIdx.x >> 5;
    int hf = lane >> 4, lm = lane & 15;
    int gid = blockIdx.x * 4 + wave;        // bh*128 + qtile
    int qt = gid & 127;
    int bh = gid >> 7;
    int b = bh >> 4, h = bh & 15;
    int q0 = qt * 16;

    // Q tile in A-layout, both 32-wide d-chunks, loaded once.
    const unsigned short* qrow = qb + ((size_t)bh * SS + q0 + lm) * HDD;
    ABPack qa0, qa1;
    qa0.u[0] = *(const U4*)(qrow + 16 * hf);
    qa0.u[1] = *(const U4*)(qrow + 16 * hf + 8);
    qa1.u[0] = *(const U4*)(qrow + 32 + 16 * hf);
    qa1.u[1] = *(const U4*)(qrow + 32 + 16 * hf + 8);

    v8f o0 = v8f_zero(), o1 = v8f_zero(), o2 = v8f_zero(), o3 = v8f_zero();
    float mi[8], li[8];
#pragma unroll
    for (int r = 0; r < 8; ++r) { mi[r] = -1e30f; li[r] = 0.f; }

    unsigned short* pw = pshare + wave * 512;
    const float scale = 0.125f;             // 1/sqrt(64)
    int nkeys = q0 + 16;                    // max allowed key+1 for this tile (block=4)

    for (int c = 0; c < nkeys; c += 32) {
        // ---- scores: S = Q * K^T, two 16-key tiles ----
        const unsigned short* k0row = kb + ((size_t)bh * SS + c + lm) * HDD;
        const unsigned short* k1row = k0row + 16 * HDD;
        ABPack b00, b01, b10, b11;
        b00.u[0] = *(const U4*)(k0row + 16 * hf);       b00.u[1] = *(const U4*)(k0row + 16 * hf + 8);
        b01.u[0] = *(const U4*)(k0row + 32 + 16 * hf);  b01.u[1] = *(const U4*)(k0row + 32 + 16 * hf + 8);
        b10.u[0] = *(const U4*)(k1row + 16 * hf);       b10.u[1] = *(const U4*)(k1row + 16 * hf + 8);
        b11.u[0] = *(const U4*)(k1row + 32 + 16 * hf);  b11.u[1] = *(const U4*)(k1row + 32 + 16 * hf + 8);

        v8f s0 = __builtin_amdgcn_wmma_f32_16x16x32_bf16(false, qa0.v, false, b00.v, (short)0, v8f_zero(), false, false);
        s0     = __builtin_amdgcn_wmma_f32_16x16x32_bf16(false, qa1.v, false, b01.v, (short)0, s0, false, false);
        v8f s1 = __builtin_amdgcn_wmma_f32_16x16x32_bf16(false, qa0.v, false, b10.v, (short)0, v8f_zero(), false, false);
        s1     = __builtin_amdgcn_wmma_f32_16x16x32_bf16(false, qa1.v, false, b11.v, (short)0, s1, false, false);

        bool need_mask = (c + 32 > q0);     // only diagonal / ragged chunks
#pragma unroll
        for (int r = 0; r < 8; ++r) {
            float a = s0[r] * scale;
            float e = s1[r] * scale;
            if (need_mask) {
                int qi = q0 + 8 * hf + r;
                if (((c + lm) >> 2)      > (qi >> 2)) a = -1e30f;
                if (((c + 16 + lm) >> 2) > (qi >> 2)) e = -1e30f;
            }
            // row max across the 16 lanes of this half-wave
            float mx = fmaxf(a, e);
            mx = fmaxf(mx, __shfl_xor(mx, 1, 32));
            mx = fmaxf(mx, __shfl_xor(mx, 2, 32));
            mx = fmaxf(mx, __shfl_xor(mx, 4, 32));
            mx = fmaxf(mx, __shfl_xor(mx, 8, 32));
            float mnew = fmaxf(mi[r], mx);
            float alpha = __expf(mi[r] - mnew);
            float p0 = __expf(a - mnew);
            float p1 = __expf(e - mnew);
            float ps = p0 + p1;
            ps += __shfl_xor(ps, 1, 32);
            ps += __shfl_xor(ps, 2, 32);
            ps += __shfl_xor(ps, 4, 32);
            ps += __shfl_xor(ps, 8, 32);
            li[r] = li[r] * alpha + ps;
            mi[r] = mnew;
            o0[r] *= alpha; o1[r] *= alpha; o2[r] *= alpha; o3[r] *= alpha;
            // stage P (C-layout -> LDS row-major 16x32)
            int prow = 8 * hf + r;
            pw[prow * 32 + lm]      = f2bf(p0);
            pw[prow * 32 + 16 + lm] = f2bf(p1);
        }
        __asm volatile("" ::: "memory");    // order DS store -> DS load (same wave; dscnt)

        // P in A-layout (16 queries x 32 keys)
        ABPack pa;
        pa.u[0] = *(const U4*)(pw + lm * 32 + 16 * hf);
        pa.u[1] = *(const U4*)(pw + lm * 32 + 16 * hf + 8);

        // ---- O += P * V : V^T rows are contiguous in key dim ----
#pragma unroll
        for (int t = 0; t < 4; ++t) {
            const unsigned short* vrow = vtb + ((size_t)bh * HDD + t * 16 + lm) * SS + c + 16 * hf;
            ABPack vb;
            vb.u[0] = *(const U4*)(vrow);
            vb.u[1] = *(const U4*)(vrow + 8);
            v8f& o = (t == 0) ? o0 : (t == 1) ? o1 : (t == 2) ? o2 : o3;
            o = __builtin_amdgcn_wmma_f32_16x16x32_bf16(false, pa.v, false, vb.v, (short)0, o, false, false);
        }
    }

    // ---- epilogue: normalize, apply sigmoid head gate, store bf16 (b,s,h*d) ----
#pragma unroll
    for (int r = 0; r < 8; ++r) {
        int row = q0 + 8 * hf + r;
        float g = gate_sig[((size_t)(b * SS + row)) * NHH + h];
        float inv = g / li[r];
        size_t base = ((size_t)(b * SS + row)) * HH + h * HDD + lm;
        ob[base]      = f2bf(o0[r] * inv);
        ob[base + 16] = f2bf(o1[r] * inv);
        ob[base + 32] = f2bf(o2[r] * inv);
        ob[base + 48] = f2bf(o3[r] * inv);
    }
}

// ---------------------------------------------------------------------------
extern "C" void kernel_launch(void* const* d_in, const int* in_sizes, int n_in,
                              void* d_out, int out_size, void* d_ws, size_t ws_size,
                              hipStream_t stream) {
    const float* x     = (const float*)d_in[0];
    const float* cosT  = (const float*)d_in[1];
    const float* sinT  = (const float*)d_in[2];
    const float* Wq    = (const float*)d_in[3];
    const float* Wk    = (const float*)d_in[4];
    const float* Wv    = (const float*)d_in[5];
    const float* Wgate = (const float*)d_in[6];
    const float* Wg    = (const float*)d_in[7];
    const float* Wo    = (const float*)d_in[8];
    float* out = (float*)d_out;

    // ---- workspace carving (~97 MB total) ----
    char* ws = (char*)d_ws;
    size_t off = 0;
    auto alloc = [&](size_t bytes) {
        void* p = ws + off;
        off += (bytes + 255) & ~(size_t)255;
        return p;
    };
    const size_t NX = (size_t)MM * HH;          // 4096*1024
    const size_t NW = (size_t)HH * HH;          // 1024*1024
    unsigned short* xb   = (unsigned short*)alloc(NX * 2);
    unsigned short* wqb  = (unsigned short*)alloc(NW * 2);
    unsigned short* wkb  = (unsigned short*)alloc(NW * 2);
    unsigned short* wvb  = (unsigned short*)alloc(NW * 2);
    unsigned short* wob  = (unsigned short*)alloc(NW * 2);
    float*          qf   = (float*)alloc(NX * 4);
    float*          kf   = (float*)alloc(NX * 4);
    float*          vf   = (float*)alloc(NX * 4);
    float*          gts  = (float*)alloc((size_t)MM * NHH * 4);
    unsigned short* qbf  = (unsigned short*)alloc(NX * 2);
    unsigned short* kbf  = (unsigned short*)alloc(NX * 2);
    unsigned short* vtb  = (unsigned short*)alloc(NX * 2);
    unsigned short* obf  = (unsigned short*)alloc(NX * 2);
    (void)ws_size; (void)in_sizes; (void)n_in; (void)out_size;

    // ---- f32 -> bf16 converts ----
    cvt_bf16_kernel<<<(int)((NX + 255) / 256), 256, 0, stream>>>(x,  xb,  (int)NX);
    cvt_bf16_kernel<<<(int)((NW + 255) / 256), 256, 0, stream>>>(Wq, wqb, (int)NW);
    cvt_bf16_kernel<<<(int)((NW + 255) / 256), 256, 0, stream>>>(Wk, wkb, (int)NW);
    cvt_bf16_kernel<<<(int)((NW + 255) / 256), 256, 0, stream>>>(Wv, wvb, (int)NW);
    cvt_bf16_kernel<<<(int)((NW + 255) / 256), 256, 0, stream>>>(Wo, wob, (int)NW);

    // ---- QKV projections (WMMA GEMM, async-LDS staged B) ----
    dim3 gg(HH / 64, MM / 64);                  // (16, 64)
    gemm_bf16_kernel<<<gg, 128, 0, stream>>>(xb, wqb, qf, MM, HH, HH);
    gemm_bf16_kernel<<<gg, 128, 0, stream>>>(xb, wkb, kf, MM, HH, HH);
    gemm_bf16_kernel<<<gg, 128, 0, stream>>>(xb, wvb, vf, MM, HH, HH);

    // ---- per-head output gate (sigmoid applied here) ----
    gate_kernel<<<MM * NHH / 4, 128, 0, stream>>>(x, Wgate, gts);

    // ---- RoPE + key gating + layout/precision change ----
    rope_gate_kernel<<<BB * NHH * SS / 4, 128, 0, stream>>>(qf, kf, vf, cosT, sinT, Wg,
                                                            qbf, kbf, vtb);

    // ---- flash attention with block-causal mask + head gate ----
    attn_kernel<<<BB * NHH * (SS / 16) / 4, 128, 0, stream>>>(qbf, kbf, vtb, gts, obf);

    // ---- output projection into d_out ----
    gemm_bf16_kernel<<<gg, 128, 0, stream>>>(obf, wob, out, MM, HH, HH);
}
